// CMVNet_27702539059341
// MI455X (gfx1250) — matrix operand order repaired and checked by reading, not compiled, not run on hardware
//
#include <hip/hip_runtime.h>
#include <hip/hip_bf16.h>
#include <stdint.h>

// Fused: h = relu(x @ W1^T + b1); y = sigmoid(dot(We[c[num]], h) + be[c[num]])
// N=131072, D_IN=256, HID=1024, OUT=1, E=100

typedef __attribute__((ext_vector_type(16))) __bf16 v16bf;
typedef __attribute__((ext_vector_type(8)))  __bf16 v8bf;
typedef __attribute__((ext_vector_type(4)))  __bf16 v4bf;
typedef __attribute__((ext_vector_type(8)))  float  v8f;

#define N_TOT 131072
#define D_IN  256
#define HID   1024
#define ROWS  64      // rows of x per workgroup
#define BLOCK 256     // 8 waves (wave32)

// ---- One-time W1 f32 -> bf16 conversion (W1 reused by all 2048 blocks) ----
__global__ __launch_bounds__(256)
void w1_to_bf16_kernel(const float* __restrict__ W1, __bf16* __restrict__ W1bf)
{
    const int i = (blockIdx.x * 256 + threadIdx.x) * 4;
    const float4 f = *reinterpret_cast<const float4*>(W1 + i);
    v4bf v;
    v[0] = (__bf16)f.x; v[1] = (__bf16)f.y;
    v[2] = (__bf16)f.z; v[3] = (__bf16)f.w;
    *reinterpret_cast<v4bf*>(W1bf + i) = v;
}

template<bool USE_BF>
__global__ __launch_bounds__(BLOCK)
void moe_fused_kernel(const float*  __restrict__ x,
                      const float*  __restrict__ W1f,
                      const __bf16* __restrict__ W1bf,
                      const float*  __restrict__ b1,
                      const float*  __restrict__ We,
                      const float*  __restrict__ be,
                      const int*    __restrict__ num,
                      const int*    __restrict__ cexp,
                      float* __restrict__ out)
{
    __shared__ float  xstage[ROWS * D_IN];  // 64 KB: raw f32 x tile (async landing zone)
    __shared__ __bf16 xs[ROWS * D_IN];      // 32 KB: x tile in bf16, row-major
    __shared__ float  accs[ROWS];           // per-row partial dot accumulators
    __shared__ int    eIdx[ROWS];           // per-row expert index c[num[row]]

    const int tid     = threadIdx.x;
    const int wave    = tid >> 5;
    const int lane    = tid & 31;
    const int hl      = lane >> 4;          // lane half (0/1), wave32
    const int ln      = lane & 15;          // lane-in-half
    const int rowBase = blockIdx.x * ROWS;

    if (tid < ROWS) {
        const int r = rowBase + tid;
        eIdx[tid] = cexp[num[r]];
        accs[tid] = 0.0f;
    }

    // ---- Async-copy x tile (f32) global -> LDS staging (ASYNCcnt path) ----
    const int rr = tid >> 2;                // 0..63 : row within tile
    const int c0 = (tid & 3) * 64;          // 0,64,128,192 : col chunk
    {
        const float* gsrc = x + (size_t)(rowBase + rr) * D_IN + c0;
        const unsigned lbase =
            (unsigned)(uintptr_t)(&xstage[rr * D_IN + c0]);
        #pragma unroll
        for (int i = 0; i < 16; ++i) {
            unsigned           ldsa = lbase + (unsigned)(i * 16);
            unsigned long long ga   = (unsigned long long)(uintptr_t)gsrc
                                      + (unsigned long long)(i * 16);
            asm volatile("global_load_async_to_lds_b128 %0, %1, off"
                         :: "v"(ldsa), "v"(ga) : "memory");
        }
        asm volatile("s_wait_asynccnt 0" ::: "memory");
    }
    __syncthreads();

    // ---- Convert staged f32 tile -> bf16 tile (one pass) ----
    #pragma unroll
    for (int i = 0; i < 16; ++i) {
        const float4 f =
            *reinterpret_cast<const float4*>(&xstage[rr * D_IN + c0 + i * 4]);
        v4bf v;
        v[0] = (__bf16)f.x; v[1] = (__bf16)f.y;
        v[2] = (__bf16)f.z; v[3] = (__bf16)f.w;
        *reinterpret_cast<v4bf*>(&xs[rr * D_IN + c0 + i * 4]) = v;
    }

    // Hoist per-lane expert weight-row element bases (32-bit offsets: e*HID < 400K)
    unsigned wbase[4][8];
    #pragma unroll
    for (int mt = 0; mt < 4; ++mt)
        #pragma unroll
        for (int r = 0; r < 8; ++r)
            wbase[mt][r] = (unsigned)eIdx[mt * 16 + r + 8 * hl] * HID;

    __builtin_prefetch(W1f + (size_t)(wave * 128) * D_IN, 0, 1);
    __syncthreads();

    // Per-lane running partials: position (m = mt*16 + r + 8*hl, n = ln) is
    // fixed across all nt iterations, so reduce across lanes only once at end.
    float part[4][8] = {};

    // ---- Each wave owns HID columns [wave*128, wave*128+128): 8 N-tiles ----
    #pragma unroll 1
    for (int nt = 0; nt < 8; ++nt) {
        const int col = wave * 128 + nt * 16 + ln;   // this lane's HID column
        v8f acc[4] = {};                             // 4 M-tiles of 16 rows

        #pragma unroll 2
        for (int k0 = 0; k0 < D_IN; k0 += 32) {
            // B fragment (K x N = 32 x 16 bf16): lane(n=ln, half=hl),
            // element j holds K = k0 + hl*16 + j -> 16 contiguous elems of W1 row
            v16bf bf;
            if constexpr (USE_BF) {
                union { v16bf v; v8bf h[2]; } bu;
                const v8bf* wp = reinterpret_cast<const v8bf*>(
                    W1bf + (size_t)col * D_IN + k0 + hl * 16);
                bu.h[0] = wp[0];
                bu.h[1] = wp[1];
                bf = bu.v;
            } else {
                const float4* wp = reinterpret_cast<const float4*>(
                    W1f + (size_t)col * D_IN + k0 + hl * 16);
                #pragma unroll
                for (int i = 0; i < 4; ++i) {
                    float4 f = wp[i];
                    bf[4 * i + 0] = (__bf16)f.x; bf[4 * i + 1] = (__bf16)f.y;
                    bf[4 * i + 2] = (__bf16)f.z; bf[4 * i + 3] = (__bf16)f.w;
                }
            }

            #pragma unroll
            for (int mt = 0; mt < 4; ++mt) {
                // A fragment (M x K = 16 x 32 bf16): lane m = mt*16 + ln,
                // element j holds K = k0 + hl*16 + j
                union { v16bf v; v8bf h[2]; } af;
                const int m    = mt * 16 + ln;
                const int base = m * D_IN + k0 + hl * 16;
                af.h[0] = *reinterpret_cast<const v8bf*>(&xs[base]);
                af.h[1] = *reinterpret_cast<const v8bf*>(&xs[base + 8]);

                acc[mt] = __builtin_amdgcn_wmma_f32_16x16x32_bf16(
                    false, af.v, false, bf, (short)0, acc[mt], false, false);
            }
        }

        // ---- Epilogue: bias + ReLU + expert-weight product into partials ----
        const float bias = b1[col];
        #pragma unroll
        for (int mt = 0; mt < 4; ++mt) {
            #pragma unroll
            for (int r = 0; r < 8; ++r) {
                float h = acc[mt][r] + bias;
                h = h > 0.0f ? h : 0.0f;
                part[mt][r] += h * We[wbase[mt][r] + (unsigned)col];
            }
        }
    }

    // ---- One cross-lane reduction per output row, then one LDS atomic ----
    #pragma unroll
    for (int mt = 0; mt < 4; ++mt) {
        #pragma unroll
        for (int r = 0; r < 8; ++r) {
            float p = part[mt][r];
            #pragma unroll
            for (int off = 8; off >= 1; off >>= 1)
                p += __shfl_xor(p, off, 16);
            if (ln == 0)
                atomicAdd(&accs[mt * 16 + r + 8 * hl], p);
        }
    }

    __syncthreads();

    if (tid < ROWS) {
        const int e = eIdx[tid];
        const float y = accs[tid] + be[e];
        out[rowBase + tid] = 1.0f / (1.0f + __expf(-y));
    }
}

extern "C" void kernel_launch(void* const* d_in, const int* in_sizes, int n_in,
                              void* d_out, int out_size, void* d_ws, size_t ws_size,
                              hipStream_t stream) {
    const float* x   = (const float*)d_in[0];
    const float* W1  = (const float*)d_in[1];
    const float* b1  = (const float*)d_in[2];
    const float* We  = (const float*)d_in[3];
    const float* be  = (const float*)d_in[4];
    const int*   num = (const int*)d_in[5];
    const int*   c   = (const int*)d_in[6];
    float* out = (float*)d_out;

    const dim3 grid(N_TOT / ROWS);                       // 2048 workgroups
    const size_t w1bf_bytes = (size_t)HID * D_IN * 2;    // 512 KB

    if (ws_size >= w1bf_bytes) {
        __bf16* w1bf = (__bf16*)d_ws;
        w1_to_bf16_kernel<<<(HID * D_IN) / 1024, 256, 0, stream>>>(W1, w1bf);
        moe_fused_kernel<true><<<grid, BLOCK, 0, stream>>>(
            x, W1, w1bf, b1, We, be, num, c, out);
    } else {
        moe_fused_kernel<false><<<grid, BLOCK, 0, stream>>>(
            x, W1, nullptr, b1, We, be, num, c, out);
    }
}
